// GraphNet_71425306133175
// MI455X (gfx1250) — compile-verified, matrix operand order checked by reading, not run on hardware
//
#include <hip/hip_runtime.h>
#include <hip/hip_bf16.h>
#include <stdint.h>
#include <stddef.h>

typedef __bf16 bf16_t;
typedef __attribute__((ext_vector_type(16))) __bf16 v16bf;
typedef __attribute__((ext_vector_type(8)))  __bf16 v8bf;
typedef __attribute__((ext_vector_type(4)))  __bf16 v4bf;
typedef __attribute__((ext_vector_type(8)))  float  v8f;
typedef __attribute__((ext_vector_type(4)))  float  v4f;

#define NN 100000
#define NE 600000
#define LAT 128
#define NSTEPS 15
#define MT 64            // rows per workgroup tile
#define TPB 256          // 8 wave32 waves

// LDS row strides (elements); multiples of 8 for 16B-aligned v8bf loads,
// chosen so DWORD stride mod 64 != 0 (bank-conflict mitigation).
#define LDA_K384 392
#define LDA_K256 264
#define LDA_K128 136
#define LDA_K32  40
#define LDA_F32  132     // f32 staging stride for LayerNorm

struct MLPW {
    const bf16_t* w[4];   // transposed bf16 weights: [N][Kp]
    const float*  b[4];   // f32 biases
    const float*  g;      // LN gamma (or null)
    const float*  b_ln;   // LN beta  (or null)
};

// ---------------- device helpers ----------------

// native f32 -> bf16 (RNE) conversion; lowers to v_cvt_*bf16_f32 on gfx1250
__device__ __forceinline__ bf16_t f2bf(float x) { return (bf16_t)x; }

// packed convert + single 8-byte LDS store
__device__ __forceinline__ void st4(bf16_t* d, float4 v) {
    v4f f; f[0] = v.x; f[1] = v.y; f[2] = v.z; f[3] = v.w;
    *(v4bf*)d = __builtin_convertvector(f, v4bf);
}

// 16x32 bf16 fragment per ISA 7.12.2: lane m=lane&15, half-wave g picks K group.
// halves 0..7 -> K = k0 + 8g + i ; halves 8..15 -> K = k0 + 16 + 8g + i
__device__ __forceinline__ v16bf frag_load(const bf16_t* rowk, int g) {
    v8bf lo = *(const v8bf*)(rowk + 8 * g);
    v8bf hi = *(const v8bf*)(rowk + 16 + 8 * g);
    v16bf r;
#pragma unroll
    for (int i = 0; i < 8; ++i) { r[i] = lo[i]; r[i + 8] = hi[i]; }
    return r;
}

template<int KP, int NT>
__device__ __forceinline__ void gemm_acc(const bf16_t* A, int lda,
                                         const bf16_t* Wt,      // [N][KP]
                                         int mrow, int g, int n0, int l15,
                                         v8f* acc) {
#pragma unroll
    for (int k0 = 0; k0 < KP; k0 += 32) {
        v16bf a = frag_load(A + (size_t)mrow * lda + k0, g);
#pragma unroll
        for (int nt = 0; nt < NT; ++nt) {
            v16bf b = frag_load(Wt + (size_t)(n0 + 16 * nt + l15) * KP + k0, g);
            acc[nt] = __builtin_amdgcn_wmma_f32_16x16x32_bf16(
                false, a, false, b, (short)0, acc[nt], false, false);
        }
    }
}

// D layout per ISA: VGPR r, lane l: N=l&15, M=r+8*(l>>4)
template<int NT>
__device__ __forceinline__ void epi_bf16(bf16_t* Aout, int lda2, const float* bias,
                                         bool relu, int mb, int nb, int lane,
                                         const v8f* acc) {
    int g = lane >> 4, l15 = lane & 15;
#pragma unroll
    for (int nt = 0; nt < NT; ++nt) {
        int n = nb + 16 * nt + l15;
        float bv = bias[n];
#pragma unroll
        for (int r = 0; r < 8; ++r) {
            int m = mb + 8 * g + r;
            float v = acc[nt][r] + bv;
            if (relu) v = fmaxf(v, 0.f);
            Aout[(size_t)m * lda2 + n] = f2bf(v);
        }
    }
}

template<int NT>
__device__ __forceinline__ void epi_f32(float* Fo, int lda2, const float* bias,
                                        int mb, int nb, int lane, const v8f* acc) {
    int g = lane >> 4, l15 = lane & 15;
#pragma unroll
    for (int nt = 0; nt < NT; ++nt) {
        int n = nb + 16 * nt + l15;
        float bv = bias[n];
#pragma unroll
        for (int r = 0; r < 8; ++r) {
            int m = mb + 8 * g + r;
            Fo[(size_t)m * lda2 + n] = acc[nt][r] + bv;
        }
    }
}

// Run 4 layers: KP0 -> 128 -> 128 -> 128 (ReLU on first 3), final stored f32.
template<int KP0, int LDA0>
__device__ __forceinline__ void mlp4_to_f32(char* smem, const MLPW& P, int tid) {
    bf16_t* A = (bf16_t*)smem;
    float*  F = (float*)smem;
    int lane = tid & 31, wv = tid >> 5;
    int mb = 16 * (wv >> 1), nb = 64 * (wv & 1);
    int l15 = lane & 15, g = lane >> 4, mrow = mb + l15;
    {
        v8f acc[4] = {};
        gemm_acc<KP0, 4>(A, LDA0, P.w[0], mrow, g, nb, l15, acc);
        __syncthreads();
        epi_bf16<4>(A, LDA_K128, P.b[0], true, mb, nb, lane, acc);
        __syncthreads();
    }
    {
        v8f acc[4] = {};
        gemm_acc<128, 4>(A, LDA_K128, P.w[1], mrow, g, nb, l15, acc);
        __syncthreads();
        epi_bf16<4>(A, LDA_K128, P.b[1], true, mb, nb, lane, acc);
        __syncthreads();
    }
    {
        v8f acc[4] = {};
        gemm_acc<128, 4>(A, LDA_K128, P.w[2], mrow, g, nb, l15, acc);
        __syncthreads();
        epi_bf16<4>(A, LDA_K128, P.b[2], true, mb, nb, lane, acc);
        __syncthreads();
    }
    {
        v8f acc[4] = {};
        gemm_acc<128, 4>(A, LDA_K128, P.w[3], mrow, g, nb, l15, acc);
        __syncthreads();
        epi_f32<4>(F, LDA_F32, P.b[3], mb, nb, lane, acc);
        __syncthreads();
    }
}

__device__ __forceinline__ void ln_stats(const float* R, float& mean, float& inv) {
    float s = 0.f, s2 = 0.f;
#pragma unroll
    for (int c = 0; c < LAT; c += 4) {
        float4 v = *(const float4*)(R + c);
        s  += v.x + v.y + v.z + v.w;
        s2 += v.x * v.x + v.y * v.y + v.z * v.z + v.w * v.w;
    }
    mean = s * (1.f / 128.f);
    float var = s2 * (1.f / 128.f) - mean * mean;
    inv = rsqrtf(var + 1e-5f);
}

// ---------------- kernels ----------------

__global__ void k_cvtw(const float* __restrict__ W, bf16_t* __restrict__ out,
                       int K, int N, int Kp, int total) {
    int idx = blockIdx.x * 256 + threadIdx.x;
    if (idx < total) {
        int n = idx / Kp, k = idx % Kp;
        float v = (n < N && k < K) ? W[(size_t)k * N + n] : 0.f;
        out[idx] = f2bf(v);
    }
}

__global__ void k_zero_i32(int* p, int n) {
    int i = blockIdx.x * 256 + threadIdx.x;
    if (i < n) p[i] = 0;
}

__global__ void k_count(const int* __restrict__ ei, int* counts) {
    int i = blockIdx.x * 256 + threadIdx.x;
    if (i < NE) atomicAdd(&counts[ei[NE + i]], 1);
}

__global__ void k_scan(const int* __restrict__ counts, int* __restrict__ offs) {
    __shared__ int sh[1024];
    __shared__ int carry;
    int tid = threadIdx.x;
    if (tid == 0) carry = 0;
    __syncthreads();
    for (int base = 0; base < NN; base += 1024) {
        int idx = base + tid;
        int v = (idx < NN) ? counts[idx] : 0;
        sh[tid] = v; __syncthreads();
        for (int o = 1; o < 1024; o <<= 1) {
            int t = (tid >= o) ? sh[tid - o] : 0;
            __syncthreads();
            sh[tid] += t;
            __syncthreads();
        }
        if (idx < NN) offs[idx] = carry + sh[tid] - v;
        __syncthreads();
        if (tid == 0) carry += sh[1023];
        __syncthreads();
    }
    if (tid == 0) offs[NN] = carry;
}

__global__ void k_fill(const int* __restrict__ ei, const int* __restrict__ offs,
                       int* cursor, int* csr) {
    int i = blockIdx.x * 256 + threadIdx.x;
    if (i < NE) {
        int d = ei[NE + i];
        int slot = atomicAdd(&cursor[d], 1);
        csr[offs[d] + slot] = i;
    }
}

__global__ void k_sort(const int* __restrict__ offs, int* csr) {
    int n = blockIdx.x * 256 + threadIdx.x;
    if (n < NN) {
        int a = offs[n], b = offs[n + 1];
        for (int i = a + 1; i < b; ++i) {
            int key = csr[i]; int j = i - 1;
            while (j >= a && csr[j] > key) { csr[j + 1] = csr[j]; --j; }
            csr[j + 1] = key;
        }
    }
}

// Encoder: feat [M,F] -> MLP(4) -> LN -> out [M,128]
__global__ void k_encoder(const float* __restrict__ feat, int F, int M,
                          MLPW P, float* __restrict__ out) {
    extern __shared__ char smem[];
    bf16_t* A = (bf16_t*)smem;
    int tid = threadIdx.x;
    int tile = blockIdx.x * MT;
    {
        int rl = tid >> 2, p = tid & 3;
        int row = tile + rl;
        bf16_t* Ar = A + (size_t)rl * LDA_K32;
#pragma unroll
        for (int i = 0; i < 8; ++i) {
            int col = 8 * p + i;
            float v = (row < M && col < F) ? feat[(size_t)row * F + col] : 0.f;
            Ar[col] = f2bf(v);
        }
    }
    __syncthreads();
    mlp4_to_f32<32, LDA_K32>(smem, P, tid);
    if (tid < MT) {
        int row = tile + tid;
        if (row < M) {
            const float* R = (const float*)smem + (size_t)tid * LDA_F32;
            float mean, inv; ln_stats(R, mean, inv);
            float* O = out + (size_t)row * LAT;
#pragma unroll
            for (int c = 0; c < LAT; c += 4) {
                float4 v = *(const float4*)(R + c);
                float4 gg = *(const float4*)(P.g + c);
                float4 bb = *(const float4*)(P.b_ln + c);
                float4 u;
                u.x = (v.x - mean) * inv * gg.x + bb.x;
                u.y = (v.y - mean) * inv * gg.y + bb.y;
                u.z = (v.z - mean) * inv * gg.z + bb.z;
                u.w = (v.w - mean) * inv * gg.w + bb.w;
                *(float4*)(O + c) = u;
            }
        }
    }
}

// Edge block: gather [h[dst]|h[src]|e] -> MLP(4) -> LN -> e_upd, e += e_upd
__global__ void k_edge(const float* __restrict__ h,
                       float* __restrict__ e,
                       float* __restrict__ e_upd,
                       const int* __restrict__ ei,
                       MLPW P) {
    extern __shared__ char smem[];
    bf16_t* A = (bf16_t*)smem;
    int tid = threadIdx.x;
    int tile = blockIdx.x * MT;          // NE divisible by MT
    {
        int el = tid >> 2, p = tid & 3;
        int edge = tile + el;
        int srcn = ei[edge];
        int dstn = ei[NE + edge];
        bf16_t* Ar = A + (size_t)el * LDA_K384;
        int c0 = 32 * p;
        const float4* hd = (const float4*)(h + (size_t)dstn * LAT) + 8 * p;
        const float4* hs = (const float4*)(h + (size_t)srcn * LAT) + 8 * p;
        const float4* ep = (const float4*)(e + (size_t)edge * LAT) + 8 * p;
#pragma unroll
        for (int i = 0; i < 8; ++i) st4(Ar + c0 + 4 * i, hd[i]);
#pragma unroll
        for (int i = 0; i < 8; ++i) st4(Ar + 128 + c0 + 4 * i, hs[i]);
#pragma unroll
        for (int i = 0; i < 8; ++i) st4(Ar + 256 + c0 + 4 * i, ep[i]);
    }
    __syncthreads();
    mlp4_to_f32<384, LDA_K384>(smem, P, tid);
    if (tid < MT) {
        int row = tile + tid;
        const float* R = (const float*)smem + (size_t)tid * LDA_F32;
        float mean, inv; ln_stats(R, mean, inv);
        float* EU = e_upd + (size_t)row * LAT;
        float* EO = e + (size_t)row * LAT;
#pragma unroll
        for (int c = 0; c < LAT; c += 4) {
            float4 v = *(const float4*)(R + c);
            float4 gg = *(const float4*)(P.g + c);
            float4 bb = *(const float4*)(P.b_ln + c);
            float4 u;
            u.x = (v.x - mean) * inv * gg.x + bb.x;
            u.y = (v.y - mean) * inv * gg.y + bb.y;
            u.z = (v.z - mean) * inv * gg.z + bb.z;
            u.w = (v.w - mean) * inv * gg.w + bb.w;
            *(float4*)(EU + c) = u;
            float4 o = *(const float4*)(EO + c);
            o.x += u.x; o.y += u.y; o.z += u.z; o.w += u.w;
            *(float4*)(EO + c) = o;
        }
    }
}

// Node block: deterministic CSR gather-sum of e_upd, [h|agg] -> MLP(4) -> LN -> h +=
__global__ void k_node(float* __restrict__ h,
                       const float* __restrict__ e_upd,
                       const int* __restrict__ offs,
                       const int* __restrict__ csr,
                       MLPW P) {
    extern __shared__ char smem[];
    bf16_t* A = (bf16_t*)smem;
    int tid = threadIdx.x;
    int tile = blockIdx.x * MT;
    {
        int nl = tid >> 2, p = tid & 3;
        int node = tile + nl;
        bf16_t* Ar = A + (size_t)nl * LDA_K256;
        int c0 = 32 * p;
        if (node < NN) {
            const float4* hr = (const float4*)(h + (size_t)node * LAT) + 8 * p;
#pragma unroll
            for (int i = 0; i < 8; ++i) st4(Ar + c0 + 4 * i, hr[i]);
            float4 acc[8];
#pragma unroll
            for (int i = 0; i < 8; ++i) acc[i] = make_float4(0.f, 0.f, 0.f, 0.f);
            int a = offs[node], bnd = offs[node + 1];
            for (int k = a; k < bnd; ++k) {
                int eg = csr[k];
                const float4* up = (const float4*)(e_upd + (size_t)eg * LAT) + 8 * p;
#pragma unroll
                for (int i = 0; i < 8; ++i) {
                    float4 v = up[i];
                    acc[i].x += v.x; acc[i].y += v.y; acc[i].z += v.z; acc[i].w += v.w;
                }
            }
#pragma unroll
            for (int i = 0; i < 8; ++i) st4(Ar + 128 + c0 + 4 * i, acc[i]);
        } else {
            float4 z = make_float4(0.f, 0.f, 0.f, 0.f);
#pragma unroll
            for (int i = 0; i < 8; ++i) { st4(Ar + c0 + 4 * i, z); st4(Ar + 128 + c0 + 4 * i, z); }
        }
    }
    __syncthreads();
    mlp4_to_f32<256, LDA_K256>(smem, P, tid);
    if (tid < MT) {
        int row = tile + tid;
        if (row < NN) {
            const float* R = (const float*)smem + (size_t)tid * LDA_F32;
            float mean, inv; ln_stats(R, mean, inv);
            float* H = h + (size_t)row * LAT;
#pragma unroll
            for (int c = 0; c < LAT; c += 4) {
                float4 v = *(const float4*)(R + c);
                float4 gg = *(const float4*)(P.g + c);
                float4 bb = *(const float4*)(P.b_ln + c);
                float4 o = *(const float4*)(H + c);
                o.x += (v.x - mean) * inv * gg.x + bb.x;
                o.y += (v.y - mean) * inv * gg.y + bb.y;
                o.z += (v.z - mean) * inv * gg.z + bb.z;
                o.w += (v.w - mean) * inv * gg.w + bb.w;
                *(float4*)(H + c) = o;
            }
        }
    }
}

// Decoder: h -> 3x(128 ReLU) -> 128x3 (N padded to 16) -> out [N,3]
__global__ void k_dec(const float* __restrict__ h, MLPW P, float* __restrict__ out) {
    extern __shared__ char smem[];
    bf16_t* A = (bf16_t*)smem;
    int tid = threadIdx.x;
    int tile = blockIdx.x * MT;
    {
        int rl = tid >> 2, p = tid & 3;
        int row = tile + rl;
        bf16_t* Ar = A + (size_t)rl * LDA_K128;
        int c0 = 32 * p;
        if (row < NN) {
            const float4* hr = (const float4*)(h + (size_t)row * LAT) + 8 * p;
#pragma unroll
            for (int i = 0; i < 8; ++i) st4(Ar + c0 + 4 * i, hr[i]);
        } else {
            float4 z = make_float4(0.f, 0.f, 0.f, 0.f);
#pragma unroll
            for (int i = 0; i < 8; ++i) st4(Ar + c0 + 4 * i, z);
        }
    }
    __syncthreads();
    int lane = tid & 31, wv = tid >> 5;
    int mb = 16 * (wv >> 1), nb = 64 * (wv & 1);
    int l15 = lane & 15, g = lane >> 4, mrow = mb + l15;
#pragma unroll
    for (int layer = 0; layer < 3; ++layer) {
        v8f acc[4] = {};
        gemm_acc<128, 4>(A, LDA_K128, P.w[layer], mrow, g, nb, l15, acc);
        __syncthreads();
        epi_bf16<4>(A, LDA_K128, P.b[layer], true, mb, nb, lane, acc);
        __syncthreads();
    }
    if ((wv & 1) == 0) {                       // 4 waves handle the N=16 (3 valid) tile
        v8f acc1[1] = {};
        gemm_acc<128, 1>(A, LDA_K128, P.w[3], mrow, g, 0, l15, acc1);
        if (l15 < 3) {
            float bv = P.b[3][l15];
#pragma unroll
            for (int r = 0; r < 8; ++r) {
                int row = tile + mb + 8 * g + r;
                if (row < NN) out[(size_t)row * 3 + l15] = acc1[0][r] + bv;
            }
        }
    }
}

// ---------------- host ----------------

extern "C" void kernel_launch(void* const* d_in, const int* in_sizes, int n_in,
                              void* d_out, int out_size, void* d_ws, size_t ws_size,
                              hipStream_t stream) {
    (void)in_sizes; (void)n_in; (void)out_size;
    const float* x         = (const float*)d_in[0];
    const float* edge_attr = (const float*)d_in[1];
    const int*   ei        = (const int*)d_in[2];     // int32 (JAX x64 off)
    auto F = [&](int i) { return (const float*)d_in[i]; };

    char* ws = (char*)d_ws;
    size_t off = 0;
    auto alloc = [&](size_t bytes) { size_t o = off; off += (bytes + 255) & ~(size_t)255; return o; };
    float*  e      = (float*)(ws + alloc((size_t)NE * LAT * 4));
    float*  e_upd  = (float*)(ws + alloc((size_t)NE * LAT * 4));
    float*  h      = (float*)(ws + alloc((size_t)NN * LAT * 4));
    int*    counts = (int*)(ws + alloc((size_t)NN * 4));
    int*    cursor = (int*)(ws + alloc((size_t)NN * 4));
    int*    offs   = (int*)(ws + alloc((size_t)(NN + 1) * 4));
    int*    csr    = (int*)(ws + alloc((size_t)NE * 4));
    bf16_t* wt     = (bf16_t*)(ws + alloc((size_t)3000000 * 2));
    if (off > ws_size) return;   // workspace too small: bail deterministically

    // ---- weight conversion: transpose to [N][Kp] bf16 with zero padding ----
    size_t wcur = 0;
    auto cvt = [&](const float* W, int K, int N, int Kp, int Np) -> const bf16_t* {
        bf16_t* dst = wt + wcur;
        int total = Kp * Np;
        k_cvtw<<<(total + 255) / 256, 256, 0, stream>>>(W, dst, K, N, Kp, total);
        wcur += (size_t)total;
        return dst;
    };
    auto mlpw = [&](int idx, int K0, int Kp0, int Nlast, int Nplast,
                    const float* lng, const float* lnb) -> MLPW {
        MLPW P;
        P.w[0] = cvt(F(idx + 0), K0, 128, Kp0, 128);      P.b[0] = F(idx + 1);
        P.w[1] = cvt(F(idx + 2), 128, 128, 128, 128);     P.b[1] = F(idx + 3);
        P.w[2] = cvt(F(idx + 4), 128, 128, 128, 128);     P.b[2] = F(idx + 5);
        P.w[3] = cvt(F(idx + 6), 128, Nlast, 128, Nplast); P.b[3] = F(idx + 7);
        P.g = lng; P.b_ln = lnb;
        return P;
    };

    // param flattening (dict insertion order)
    MLPW Penc_n = mlpw(3, 11, 32, 128, 128, F(11), F(12));
    MLPW Penc_e = mlpw(13, 7, 32, 128, 128, F(21), F(22));
    MLPW Pdec   = mlpw(23, 128, 128, 3, 16, nullptr, nullptr);
    MLPW Pedge[NSTEPS], Pnode[NSTEPS];
    for (int s = 0; s < NSTEPS; ++s) {
        int bi = 31 + 20 * s;
        Pedge[s] = mlpw(bi, 384, 384, 128, 128, F(bi + 8), F(bi + 9));
        Pnode[s] = mlpw(bi + 10, 256, 256, 128, 128, F(bi + 18), F(bi + 19));
    }

    // ---- deterministic CSR (sorted edge-ids per destination node) ----
    k_zero_i32<<<(NN + 255) / 256, 256, 0, stream>>>(counts, NN);
    k_zero_i32<<<(NN + 255) / 256, 256, 0, stream>>>(cursor, NN);
    k_count<<<(NE + 255) / 256, 256, 0, stream>>>(ei, counts);
    k_scan<<<1, 1024, 0, stream>>>(counts, offs);
    k_fill<<<(NE + 255) / 256, 256, 0, stream>>>(ei, offs, cursor, csr);
    k_sort<<<(NN + 255) / 256, 256, 0, stream>>>(offs, csr);

    const int STD_LDS  = MT * LDA_F32 * 4;        // 33792 B (covers bf16 + f32 stages)
    const int EDGE_LDS = MT * LDA_K384 * 2;       // 50176 B
    const int DEC_LDS  = MT * LDA_K128 * 2;       // 17408 B

    // ---- encoders ----
    k_encoder<<<(NN + MT - 1) / MT, TPB, STD_LDS, stream>>>(x, 11, NN, Penc_n, h);
    k_encoder<<<(NE + MT - 1) / MT, TPB, STD_LDS, stream>>>(edge_attr, 7, NE, Penc_e, e);

    // ---- 15 GraphNet blocks ----
    for (int s = 0; s < NSTEPS; ++s) {
        k_edge<<<NE / MT, TPB, EDGE_LDS, stream>>>(h, e, e_upd, ei, Pedge[s]);
        k_node<<<(NN + MT - 1) / MT, TPB, STD_LDS, stream>>>(h, e_upd, offs, csr, Pnode[s]);
    }

    // ---- decoder ----
    k_dec<<<(NN + MT - 1) / MT, TPB, DEC_LDS, stream>>>(h, Pdec, (float*)d_out);
}
